// TopKRouter_11948599018367
// MI455X (gfx1250) — matrix operand order repaired and checked by reading, not compiled, not run on hardware
//
#include <hip/hip_runtime.h>
#include <hip/hip_bf16.h>
#include <math.h>

// ---------------- problem constants ----------------
#define DMODEL   2048
#define NEXP     64
#define NTOK     16384            // 4 * 4096
#define TOK_BLK  64               // tokens per workgroup (4 waves x 16)
#define LDSTRIDE 68               // 64 + pad to dodge LDS bank conflicts

// ws layout: [0,512) bytes: 128 f32 expert accumulators (psum, count)
//            [512, 512+256K): W_hi bf16, swizzled B-tile order
//            [512+256K, 512+512K): W_lo bf16, swizzled B-tile order
#define WS_ACC_FLOATS  (2 * NEXP)
#define WS_W_OFF       512        // bytes; keeps bf16 arrays 32B-aligned

typedef __attribute__((ext_vector_type(16))) __bf16 bf16x16;
typedef __attribute__((ext_vector_type(16))) float  floatx16;
typedef __attribute__((ext_vector_type(8)))  float  floatx8;

// Split one f32 vector into bf16 hi (RNE) and bf16 lo (residual).
__device__ __forceinline__ void split_bf16(floatx16 v, bf16x16& hi, bf16x16& lo) {
#pragma unroll
    for (int i = 0; i < 16; ++i) {
        __bf16 h = (__bf16)v[i];
        hi[i] = h;
        lo[i] = (__bf16)(v[i] - (float)h);
    }
}

// One-shot prep: zero accumulators and split W into bf16 hi/lo, stored
// pre-swizzled into this file's dense-B (32x16 bf16) lane order (contiguous k, as A uses):
//   dst[(g*64 + kchunk)*512 + lane*16 + j],  lane = half*16 + (e&15),
//   k = kchunk*32 + half*16 + j, e = g*16 + (e&15).
// So each wave's per-group B fetch is 32 lanes x 32B fully contiguous.
__global__ __launch_bounds__(256) void router_prep(const float* __restrict__ W,
                                                   float* __restrict__ ws) {
    int idx = blockIdx.x * 256 + threadIdx.x;       // 0 .. NEXP*DMODEL-1
    if (idx < WS_ACC_FLOATS) ws[idx] = 0.0f;        // zero atomic accumulators

    int e = idx >> 11;                               // / DMODEL
    int k = idx & (DMODEL - 1);
    float v = W[idx];
    __bf16 h = (__bf16)v;
    __bf16 l = (__bf16)(v - (float)h);

    int g    = e >> 4;
    int l15  = e & 15;
    int c    = k >> 5;           // 32-wide k chunk
    int half = (k >> 4) & 1;
    int j    = k & 15;
    int lane = half * 16 + l15;
    int dst  = (g * 64 + c) * 512 + lane * 16 + j;

    __bf16* whi = (__bf16*)((char*)ws + WS_W_OFF);
    __bf16* wlo = whi + NEXP * DMODEL;
    whi[dst] = h;
    wlo[dst] = l;
}

// Main router kernel: logits via split-bf16 WMMA, softmax/top-2 via LDS.
__global__ __launch_bounds__(128) void router_main(const float* __restrict__ x,
                                                   const __bf16* __restrict__ whi,
                                                   const __bf16* __restrict__ wlo,
                                                   float* __restrict__ out,
                                                   float* __restrict__ ws) {
    __shared__ float s_log[TOK_BLK * LDSTRIDE];
    __shared__ float s_max[TOK_BLK];
    __shared__ float s_invsum[TOK_BLK];
    __shared__ int   s_idx[TOK_BLK * 2];

    const int tid  = threadIdx.x;
    const int wave = tid >> 5;
    const int lane = tid & 31;
    const int half = lane >> 4;
    const int l15  = lane & 15;

    const int tokBase = blockIdx.x * TOK_BLK + wave * 16;

    // A (16x32 bf16) lane layout used here (contiguous k, matching B): lanes 0-15 hold row M=l15, K {0..15};
    // lanes 16-31 hold row M=l15, K {16..31}.
    const float* xrow = x + (size_t)(tokBase + l15) * DMODEL;
    const int aOff = half * 16;

    // 8 accumulator chains: [parity*4 + group]. Two parities (even/odd 32-K
    // chunk) split the K loop; each chunk issues 3 dependent WMMAs per accumulator;
    // merged once after the K loop.
    floatx8 acc[8];
#pragma unroll
    for (int g = 0; g < 8; ++g) acc[g] = (floatx8){};

#pragma unroll 1
    for (int k0 = 0; k0 < DMODEL; k0 += 64) {       // two 32-K chunks per body
#pragma unroll
        for (int p = 0; p < 2; ++p) {
            const int kc = k0 + p * 32;
            // --- A tile: 16 f32 per lane (two 32B chunks), split hi/lo ---
            const float* pa = xrow + kc + aOff;
            float4 a0 = *(const float4*)(pa);
            float4 a1 = *(const float4*)(pa + 4);
            float4 a2 = *(const float4*)(pa + 8);
            float4 a3 = *(const float4*)(pa + 12);
            floatx16 av = {a0.x, a0.y, a0.z, a0.w, a1.x, a1.y, a1.z, a1.w,
                           a2.x, a2.y, a2.z, a2.w, a3.x, a3.y, a3.z, a3.w};
            bf16x16 ahi, alo;
            split_bf16(av, ahi, alo);

            const int c = kc >> 5;
#pragma unroll
            for (int g = 0; g < 4; ++g) {
                // --- B tiles: pre-split, pre-swizzled; two aligned 32B loads ---
                int boff = (g * 64 + c) * 512 + lane * 16;
                bf16x16 bhi = *(const bf16x16*)(whi + boff);
                bf16x16 blo = *(const bf16x16*)(wlo + boff);

                const int a8 = p * 4 + g;
                // hi*hi + hi*lo + lo*hi ~= full f32 product; 3x bf16-WMMA
                // cost, still under the HBM roofline for this kernel.
                acc[a8] = __builtin_amdgcn_wmma_f32_16x16x32_bf16(
                    false, ahi, false, bhi, (short)0, acc[a8], false, false);
                acc[a8] = __builtin_amdgcn_wmma_f32_16x16x32_bf16(
                    false, ahi, false, blo, (short)0, acc[a8], false, false);
                acc[a8] = __builtin_amdgcn_wmma_f32_16x16x32_bf16(
                    false, alo, false, bhi, (short)0, acc[a8], false, false);
            }
        }
    }

    // merge parity chains
#pragma unroll
    for (int g = 0; g < 4; ++g) acc[g] += acc[4 + g];

    // ---- spill C tiles to LDS ----
    // C layout: VGPR j: lanes 0-15 -> row j, col l15; lanes 16-31 -> row j+8.
#pragma unroll
    for (int g = 0; g < 4; ++g) {
#pragma unroll
        for (int j = 0; j < 8; ++j) {
            int tl = wave * 16 + j + 8 * half;       // local token
            s_log[tl * LDSTRIDE + g * 16 + l15] = acc[g][j];
        }
    }
    __syncthreads();

    // ---- phase 1: per-token softmax stats + top-2 (threads 0..63) ----
    if (tid < TOK_BLK) {
        const float* row = &s_log[tid * LDSTRIDE];
        float m = row[0];
#pragma unroll 1
        for (int e = 1; e < NEXP; ++e) m = fmaxf(m, row[e]);

        float sum = 0.0f;
        float b1 = -3.402823466e38f, b2 = -3.402823466e38f;
        int   i1 = 0, i2 = 0;
#pragma unroll 1
        for (int e = 0; e < NEXP; ++e) {
            float l = row[e];
            sum += __expf(l - m);
            if (l > b1) { b2 = b1; i2 = i1; b1 = l; i1 = e; }   // ties -> lowest idx first
            else if (l > b2) { b2 = l; i2 = e; }
        }
        // normalized top-2 weights: softmax denominator cancels
        float e1 = __expf(b1 - m), e2 = __expf(b2 - m);
        float inv12 = 1.0f / (e1 + e2);

        int gt = blockIdx.x * TOK_BLK + tid;
        out[gt * 2 + 0] = e1 * inv12;
        out[gt * 2 + 1] = e2 * inv12;
        out[2 * NTOK + gt * 2 + 0] = (float)i1;   // indices emitted as f32
        out[2 * NTOK + gt * 2 + 1] = (float)i2;

        s_max[tid]    = m;
        s_invsum[tid] = 1.0f / sum;
        s_idx[tid * 2 + 0] = i1;
        s_idx[tid * 2 + 1] = i2;
    }
    __syncthreads();

    // ---- phase 2: per-expert prob sums + selection counts (threads 0..63) ----
    if (tid < NEXP) {
        float psum = 0.0f;
        int   cnt  = 0;
#pragma unroll 1
        for (int t = 0; t < TOK_BLK; ++t) {
            psum += __expf(s_log[t * LDSTRIDE + tid] - s_max[t]) * s_invsum[t];
            cnt  += (s_idx[t * 2] == tid) + (s_idx[t * 2 + 1] == tid);
        }
        atomicAdd(&ws[tid], psum);              // sum of probs per expert
        atomicAdd(&ws[NEXP + tid], (float)cnt); // selection count per expert
    }
}

// aux_loss = E * sum_i (cnt_i/N) * (psum_i/N)
__global__ __launch_bounds__(64) void router_finalize(const float* __restrict__ ws,
                                                      float* __restrict__ out) {
    __shared__ float red[NEXP];
    int e = threadIdx.x;
    const float invN = 1.0f / (float)NTOK;
    red[e] = (ws[NEXP + e] * invN) * (ws[e] * invN);
    __syncthreads();
#pragma unroll
    for (int s = 32; s > 0; s >>= 1) {
        if (e < s) red[e] += red[e + s];
        __syncthreads();
    }
    if (e == 0) out[4 * NTOK] = (float)NEXP * red[0];
}

extern "C" void kernel_launch(void* const* d_in, const int* in_sizes, int n_in,
                              void* d_out, int out_size, void* d_ws, size_t ws_size,
                              hipStream_t stream) {
    const float* x = (const float*)d_in[0];    // [4,4096,2048] f32
    const float* W = (const float*)d_in[1];    // [64,2048] f32
    float* out = (float*)d_out;                // weights | indices | aux
    float* ws  = (float*)d_ws;

    const __bf16* whi = (const __bf16*)((const char*)d_ws + WS_W_OFF);
    const __bf16* wlo = whi + NEXP * DMODEL;

    router_prep<<<(NEXP * DMODEL) / 256, 256, 0, stream>>>(W, ws);
    router_main<<<NTOK / TOK_BLK, 128, 0, stream>>>(x, whi, wlo, out, ws);
    router_finalize<<<1, 64, 0, stream>>>(ws, out);
}
